// Denoiser_77841987273333
// MI455X (gfx1250) — matrix operand order repaired and checked by reading, not compiled, not run on hardware
//
#include <hip/hip_runtime.h>
#include <hip/hip_bf16.h>

typedef _Float16 half_t;
typedef __attribute__((ext_vector_type(16))) _Float16 v16h;
typedef __attribute__((ext_vector_type(8)))  _Float16 v8h;
typedef __attribute__((ext_vector_type(8)))  float    v8f;

#define LRELU_SLOPE 0.01f

__device__ __forceinline__ float lrelu(float v) { return v >= 0.f ? v : LRELU_SLOPE * v; }

// ---- WMMA fragment helpers (CDNA5 ISA 7.12.2 layouts, wave32) ----

// A-matrix 16x32 f16 swizzle: lane = hlf*16 + m; element e -> K in chunk:
//   e in [0,8)  : K = 8*hlf + e
//   e in [8,16) : K = 16 + 8*hlf + (e-8)
__device__ __forceinline__ int a_kmap(int e, int hlf) {
    int v = e >> 1, p = e & 1;
    return ((v < 4) ? (2 * v) : (2 * v + 8)) + 8 * hlf + p;
}

// Two contiguous 16B LDS loads per fragment.
__device__ __forceinline__ v16h load_afrag(const half_t* src, int ld, int lane, int kbase) {
    int m = lane & 15, hlf = lane >> 4;
    const half_t* r = src + m * ld + kbase + 8 * hlf;
    v8h lo = *(const v8h*)r;
    v8h hi = *(const v8h*)(r + 16);
    return __builtin_shufflevector(lo, hi, 0, 1, 2, 3, 4, 5, 6, 7,
                                   8, 9, 10, 11, 12, 13, 14, 15);
}

// B-matrix 32x16 f16: lane's 16 elements are K=k0..k0+7, k0+16..k0+23 (k0=kbase+8*hlf) at fixed N -> two 16B loads.
// element (k, n) must live at base[n*sn + k]  (K innermost / contiguous).
__device__ __forceinline__ v16h load_bfrag(const half_t* base, int sn, int lane,
                                           int kbase, int nbase) {
    int n  = nbase + (lane & 15);
    int k0 = kbase + ((lane >> 4) << 3);
    const half_t* r = base + (long)n * sn + k0;
    v8h lo = *(const v8h*)r;
    v8h hi = *(const v8h*)(r + 16);
    return __builtin_shufflevector(lo, hi, 0, 1, 2, 3, 4, 5, 6, 7,
                                   8, 9, 10, 11, 12, 13, 14, 15);
}

// C/D f32 16x16: row = v + 8*(lane>>4), col = lane&15
__device__ __forceinline__ void store_cfrag_lds(half_t* dst, int ld, int lane, int nbase, v8f c) {
    int n = nbase + (lane & 15), mh = (lane >> 4) * 8;
#pragma unroll
    for (int v = 0; v < 8; ++v) dst[(mh + v) * ld + n] = (half_t)c[v];
}

__device__ __forceinline__ v8f wmma_f16(v16h a, v16h b, v8f c) {
    return __builtin_amdgcn_wmma_f32_16x16x32_f16(false, a, false, b, (short)0, c, false, false);
}

// ---- Kernel 0a: f32 -> f16 ----
__global__ void k_f32_to_f16(const float* __restrict__ src, half_t* __restrict__ dst, int n) {
    int i = blockIdx.x * 256 + threadIdx.x;
    if (i < n) dst[i] = (half_t)src[i];
}

// ---- Kernel 0b: f32 -> f16 transpose: src[o*I + i] -> dst[i*O + o] ----
__global__ void k_f32_to_f16_t(const float* __restrict__ src, half_t* __restrict__ dst,
                               int O, int I) {
    int idx = blockIdx.x * 256 + threadIdx.x;
    if (idx < O * I) {
        int o = idx / I, i = idx % I;
        dst[i * O + o] = (half_t)src[idx];
    }
}

// ---- Kernel 1: convs1 pointwise MLP 3 -> 64(relu) -> 128, store f16 ----
__global__ void __launch_bounds__(256)
k_convs1(const float* __restrict__ x, const float* __restrict__ W1, const float* __restrict__ b1,
         const float* __restrict__ W2, const float* __restrict__ b2, half_t* __restrict__ fh) {
    __shared__ float w1s[64 * 3], b1s[64], w2s[128 * 64], b2s[128];
    for (int i = threadIdx.x; i < 64 * 3; i += 256) w1s[i] = W1[i];
    for (int i = threadIdx.x; i < 64; i += 256) b1s[i] = b1[i];
    for (int i = threadIdx.x; i < 128 * 64; i += 256) w2s[i] = W2[i];
    for (int i = threadIdx.x; i < 128; i += 256) b2s[i] = b2[i];
    __syncthreads();
    long p = (long)blockIdx.x * 256 + threadIdx.x;
    float x0 = x[p * 3], x1 = x[p * 3 + 1], x2 = x[p * 3 + 2];
    float h[64];
#pragma unroll
    for (int o = 0; o < 64; ++o) {
        float a = b1s[o] + w1s[o * 3] * x0 + w1s[o * 3 + 1] * x1 + w1s[o * 3 + 2] * x2;
        h[o] = a > 0.f ? a : 0.f;
    }
    for (int o = 0; o < 128; ++o) {
        float a = b2s[o];
#pragma unroll
        for (int i = 0; i < 64; ++i) a += w2s[o * 64 + i] * h[i];
        fh[p * 128 + o] = (half_t)a;
    }
}

// ---- Kernel 2: brute-force KNN, top-17 nearest (self first), per-thread query ----
__global__ void __launch_bounds__(256)
k_knn(const float* __restrict__ x, int* __restrict__ idxb) {
    __shared__ float xt[256 * 3];
    int q = blockIdx.x * 256 + threadIdx.x;
    int base = (q >> 13) << 13;   // batch base, N = 8192
    float qx = x[(long)q * 3], qy = x[(long)q * 3 + 1], qz = x[(long)q * 3 + 2];
    float bd[17]; int bi[17];
#pragma unroll
    for (int t = 0; t < 17; ++t) { bd[t] = 3.4e38f; bi[t] = 0; }
    for (int tile = 0; tile < 8192; tile += 256) {
        __syncthreads();
        int c = tile + threadIdx.x;
        xt[threadIdx.x * 3 + 0] = x[(long)(base + c) * 3 + 0];
        xt[threadIdx.x * 3 + 1] = x[(long)(base + c) * 3 + 1];
        xt[threadIdx.x * 3 + 2] = x[(long)(base + c) * 3 + 2];
        __syncthreads();
        for (int i = 0; i < 256; ++i) {
            float dx = qx - xt[i * 3], dy = qy - xt[i * 3 + 1], dz = qz - xt[i * 3 + 2];
            float d2 = dx * dx + dy * dy + dz * dz;
            if (d2 < bd[16]) {
                bd[16] = d2; bi[16] = tile + i;
#pragma unroll
                for (int t = 16; t >= 1; --t) {
                    if (bd[t] < bd[t - 1]) {
                        float td = bd[t]; bd[t] = bd[t - 1]; bd[t - 1] = td;
                        int ti = bi[t]; bi[t] = bi[t - 1]; bi[t - 1] = ti;
                    }
                }
            }
        }
    }
#pragma unroll
    for (int t = 0; t < 17; ++t) idxb[(long)q * 17 + t] = bi[t];
}

// ---- Kernel 3: per-point WMMA chain. One wave = 16 points.
// feat0 = [f, convs2(r_self)] ; q = Wq feat0 + bq ; h = Wk^T q ; h3 = Wc3^T h2
// outputs: hout[point][0:128]=h1, [128:192]=h3 (f16); qb[point] = q.bk + h2.bc3
__global__ void __launch_bounds__(128)
k_point(const float* __restrict__ x, const half_t* __restrict__ fh,
        const half_t* __restrict__ Wc1h, const half_t* __restrict__ bc1h,
        const half_t* __restrict__ Wc2h, const half_t* __restrict__ bc2h,
        const half_t* __restrict__ Wc3h, const half_t* __restrict__ bc3h,
        const half_t* __restrict__ Wc3T, const half_t* __restrict__ Wqh,
        const half_t* __restrict__ bqh,  const half_t* __restrict__ WkT,
        const half_t* __restrict__ bkh,
        half_t* __restrict__ hout, float* __restrict__ qbout) {
    __shared__ half_t buf[4][16 * 256];
    __shared__ half_t t2s[4][16 * 64];
    __shared__ float  r9s[4][16 * 9];
    int w = threadIdx.x >> 5, lane = threadIdx.x & 31;
    int ln = lane & 15, hlf = lane >> 4;
    int p0 = (blockIdx.x * 4 + w) * 16;
    half_t* fb = buf[w];

    // prefetch hot weight lines (global_prefetch_b8)
    __builtin_prefetch(Wqh + (long)threadIdx.x * 512, 0, 1);
    __builtin_prefetch(WkT + (long)threadIdx.x * 512, 0, 1);

    // f -> cols 0..127 (128-bit copies)
    {
        const uint4* fsrc = (const uint4*)(fh + (long)p0 * 128);
        for (int c = lane; c < 16 * 16; c += 32) {
            int mm = c >> 4, d8 = c & 15;
            *(uint4*)(fb + mm * 256 + d8 * 8) = fsrc[mm * 16 + d8];
        }
    }
    // self r9 = [x, x, 0]
    if (lane < 16) {
        const float* xp = x + (long)(p0 + lane) * 3;
        float a0 = xp[0], a1 = xp[1], a2 = xp[2];
        float* r = r9s[w] + lane * 9;
        r[0] = a0; r[1] = a1; r[2] = a2; r[3] = a0; r[4] = a1; r[5] = a2;
        r[6] = 0.f; r[7] = 0.f; r[8] = 0.f;
    }
    __syncthreads();

    // t1 = lrelu(Wc1 r9 + bc1), built directly as A-fragments (K=64 -> 2 chunks)
    const float* rr = r9s[w] + ln * 9;
    v16h t1f[2];
#pragma unroll
    for (int c = 0; c < 2; ++c) {
#pragma unroll
        for (int e = 0; e < 16; ++e) {
            int k = c * 32 + a_kmap(e, hlf);
            float acc = (float)bc1h[k];
#pragma unroll
            for (int i = 0; i < 9; ++i) acc += (float)Wc1h[k * 9 + i] * rr[i];
            t1f[c][e] = (half_t)lrelu(acc);
        }
    }
    // t2 = lrelu(Wc2 t1 + bc2)  (WMMA)
    for (int t = 0; t < 4; ++t) {
        v8f c; float bias = (float)bc2h[t * 16 + ln];
#pragma unroll
        for (int v = 0; v < 8; ++v) c[v] = bias;
#pragma unroll
        for (int ch = 0; ch < 2; ++ch)
            c = wmma_f16(t1f[ch], load_bfrag(Wc2h, 64, lane, ch * 32, t * 16), c);
#pragma unroll
        for (int v = 0; v < 8; ++v) c[v] = lrelu(c[v]);
        store_cfrag_lds(t2s[w], 64, lane, t * 16, c);
    }
    __syncthreads();
    // r0 = Wc3 t2 + bc3 -> feat cols 128..255  (WMMA; B[k=c][n=o] = Wc3h[o*64+c])
    v16h t2f[2];
#pragma unroll
    for (int c = 0; c < 2; ++c) t2f[c] = load_afrag(t2s[w], 64, lane, c * 32);
    for (int t = 0; t < 8; ++t) {
        v8f c; float bias = (float)bc3h[t * 16 + ln];
#pragma unroll
        for (int v = 0; v < 8; ++v) c[v] = bias;
#pragma unroll
        for (int ch = 0; ch < 2; ++ch)
            c = wmma_f16(t2f[ch], load_bfrag(Wc3h, 64, lane, ch * 32, t * 16), c);
        store_cfrag_lds(fb + 128, 256, lane, t * 16, c);
    }
    __syncthreads();

    // q = Wq feat0 + bq  (A = feat, 8 K-chunks in regs; write q back over buf)
    v16h af[8];
#pragma unroll
    for (int c = 0; c < 8; ++c) af[c] = load_afrag(fb, 256, lane, c * 32);
    __syncthreads();
    float qbp[8];
#pragma unroll
    for (int v = 0; v < 8; ++v) qbp[v] = 0.f;
    for (int t = 0; t < 16; ++t) {
        v8f c; float bias = (float)bqh[t * 16 + ln];
#pragma unroll
        for (int v = 0; v < 8; ++v) c[v] = bias;
#pragma unroll
        for (int ch = 0; ch < 8; ++ch)
            c = wmma_f16(af[ch], load_bfrag(Wqh, 256, lane, ch * 32, t * 16), c);
        float bkv = (float)bkh[t * 16 + ln];
#pragma unroll
        for (int v = 0; v < 8; ++v) qbp[v] += c[v] * bkv;   // q . bk partial
        store_cfrag_lds(fb, 256, lane, t * 16, c);
    }
    __syncthreads();

    // h = Wk^T q : h[i] = sum_o Wk[o][i] q[o] -> B[k=o][n=i] = WkT[i*256+o] (contig)
#pragma unroll
    for (int c = 0; c < 8; ++c) af[c] = load_afrag(fb, 256, lane, c * 32);
    __syncthreads();
    for (int t = 0; t < 16; ++t) {
        v8f c;
#pragma unroll
        for (int v = 0; v < 8; ++v) c[v] = 0.f;
#pragma unroll
        for (int ch = 0; ch < 8; ++ch)
            c = wmma_f16(af[ch], load_bfrag(WkT, 256, lane, ch * 32, t * 16), c);
        if (t < 8) {      // h1 -> global
#pragma unroll
            for (int v = 0; v < 8; ++v)
                hout[(long)(p0 + hlf * 8 + v) * 192 + t * 16 + ln] = (half_t)c[v];
        } else {          // h2 -> LDS [16][128], accumulate h2 . bc3
            store_cfrag_lds(fb, 128, lane, (t - 8) * 16, c);
            float b3 = (float)bc3h[(t - 8) * 16 + ln];
#pragma unroll
            for (int v = 0; v < 8; ++v) qbp[v] += c[v] * b3;
        }
    }
    __syncthreads();

    // h3 = Wc3^T h2 : B[k=o][n=c'] = Wc3T[c'*128+o] (contig)
    v16h hf[4];
#pragma unroll
    for (int c = 0; c < 4; ++c) hf[c] = load_afrag(fb, 128, lane, c * 32);
    for (int t = 0; t < 4; ++t) {
        v8f c;
#pragma unroll
        for (int v = 0; v < 8; ++v) c[v] = 0.f;
#pragma unroll
        for (int ch = 0; ch < 4; ++ch)
            c = wmma_f16(hf[ch], load_bfrag(Wc3T, 128, lane, ch * 32, t * 16), c);
#pragma unroll
        for (int v = 0; v < 8; ++v)
            hout[(long)(p0 + hlf * 8 + v) * 192 + 128 + t * 16 + ln] = (half_t)c[v];
    }
    // qb reduce within 16-lane halves (rows v + 8*hlf)
#pragma unroll
    for (int v = 0; v < 8; ++v) {
        float s = qbp[v];
#pragma unroll
        for (int o = 1; o < 16; o <<= 1) s += __shfl_xor(s, o, 32);
        if (ln == 0) qbout[p0 + hlf * 8 + v] = s;
    }
}

// ---- Kernel 4: attention. One wave = one point, M=16 = its 16 neighbors.
__global__ void __launch_bounds__(256)
k_attn(const float* __restrict__ x, const half_t* __restrict__ fh,
       const int* __restrict__ idxb,
       const half_t* __restrict__ Wc1h, const half_t* __restrict__ bc1h,
       const half_t* __restrict__ Wc2h, const half_t* __restrict__ bc2h,
       const half_t* __restrict__ hout, const float* __restrict__ qb,
       float* __restrict__ out) {
    __shared__ float r9s[8][16 * 9];
    __shared__ float xnb[8][16 * 3];
    __shared__ float sA[8][16];
    __shared__ int   jdx[8][16];
    int w = threadIdx.x >> 5, lane = threadIdx.x & 31;
    int ln = lane & 15, hlf = lane >> 4;
    int p = blockIdx.x * 8 + w;
    int base = (p >> 13) << 13;

    if (lane < 16) {
        int j = idxb[(long)p * 17 + 1 + lane];
        jdx[w][lane] = j;
        const float* xc = x + (long)p * 3;
        const float* xj = x + (long)(base + j) * 3;
        float c0 = xc[0], c1 = xc[1], c2 = xc[2];
        float j0 = xj[0], j1 = xj[1], j2 = xj[2];
        float* r = r9s[w] + lane * 9;
        r[0] = c0; r[1] = c1; r[2] = c2; r[3] = j0; r[4] = j1; r[5] = j2;
        r[6] = c0 - j0; r[7] = c1 - j1; r[8] = c2 - j2;
        float* xn = xnb[w] + lane * 3;
        xn[0] = j0; xn[1] = j1; xn[2] = j2;
    }
    __syncthreads();

    // a1 = lrelu(Wc1 r9 + bc1) as A-fragments
    const float* rr = r9s[w] + ln * 9;
    v16h a1f[2];
#pragma unroll
    for (int c = 0; c < 2; ++c) {
#pragma unroll
        for (int e = 0; e < 16; ++e) {
            int k = c * 32 + a_kmap(e, hlf);
            float acc = (float)bc1h[k];
#pragma unroll
            for (int i = 0; i < 9; ++i) acc += (float)Wc1h[k * 9 + i] * rr[i];
            a1f[c][e] = (half_t)lrelu(acc);
        }
    }
    // a2 = lrelu(Wc2 a1 + bc2) (WMMA), dotted against h3 on the fly
    const half_t* h3p = hout + (long)p * 192 + 128;
    float part[8];
#pragma unroll
    for (int v = 0; v < 8; ++v) part[v] = 0.f;
    for (int t = 0; t < 4; ++t) {
        v8f c; float bias = (float)bc2h[t * 16 + ln];
#pragma unroll
        for (int v = 0; v < 8; ++v) c[v] = bias;
#pragma unroll
        for (int ch = 0; ch < 2; ++ch)
            c = wmma_f16(a1f[ch], load_bfrag(Wc2h, 64, lane, ch * 32, t * 16), c);
        float hv = (float)h3p[t * 16 + ln];
#pragma unroll
        for (int v = 0; v < 8; ++v) part[v] += lrelu(c[v]) * hv;
    }
#pragma unroll
    for (int v = 0; v < 8; ++v) {
        float s = part[v];
#pragma unroll
        for (int o = 1; o < 16; o <<= 1) s += __shfl_xor(s, o, 32);
        if (ln == 0) sA[w][hlf * 8 + v] = s;
    }
    // sF = h1 . f_j : lane pair (ln, ln+16) splits 128 dims; 128-bit loads
    int j = jdx[w][ln];
    const v8h* hv8 = (const v8h*)(hout + (long)p * 192 + hlf * 64);
    const v8h* fv8 = (const v8h*)(fh + (long)(base + j) * 128 + hlf * 64);
    float sf = 0.f;
#pragma unroll
    for (int d8 = 0; d8 < 8; ++d8) {
        v8h a = hv8[d8], b = fv8[d8];
#pragma unroll
        for (int e = 0; e < 8; ++e) sf += (float)a[e] * (float)b[e];
    }
    sf += __shfl_xor(sf, 16, 32);
    __syncthreads();

    // softmax over 16 neighbors (lanes 0..15), weighted sum of neighbor coords
    float s = sA[w][ln] + sf + qb[p];
    float mx = s;
#pragma unroll
    for (int o = 1; o < 16; o <<= 1) mx = fmaxf(mx, __shfl_xor(mx, o, 32));
    float e = __expf(s - mx);
    float den = e;
#pragma unroll
    for (int o = 1; o < 16; o <<= 1) den += __shfl_xor(den, o, 32);
    float wgt = e / den;
    float o0 = wgt * xnb[w][ln * 3 + 0];
    float o1 = wgt * xnb[w][ln * 3 + 1];
    float o2 = wgt * xnb[w][ln * 3 + 2];
#pragma unroll
    for (int o = 1; o < 16; o <<= 1) {
        o0 += __shfl_xor(o0, o, 32);
        o1 += __shfl_xor(o1, o, 32);
        o2 += __shfl_xor(o2, o, 32);
    }
    if (lane == 0) {
        out[(long)p * 3 + 0] = o0;
        out[(long)p * 3 + 1] = o1;
        out[(long)p * 3 + 2] = o2;
    }
}

extern "C" void kernel_launch(void* const* d_in, const int* in_sizes, int n_in,
                              void* d_out, int out_size, void* d_ws, size_t ws_size,
                              hipStream_t stream) {
    (void)in_sizes; (void)n_in; (void)out_size; (void)ws_size;
    const int P = 4 * 8192;   // 32768 points total

    char* wsb = (char*)d_ws;
    size_t off = 0;
    auto carve = [&](size_t bytes) -> void* {
        void* p = wsb + off;
        off += (bytes + 255) & ~(size_t)255;
        return p;
    };
    half_t* fh   = (half_t*)carve((size_t)P * 128 * sizeof(half_t));
    int*    idxb = (int*)   carve((size_t)P * 17 * sizeof(int));
    half_t* hout = (half_t*)carve((size_t)P * 192 * sizeof(half_t));
    float*  qbv  = (float*) carve((size_t)P * sizeof(float));
    half_t* Wc1h = (half_t*)carve(576 * sizeof(half_t));
    half_t* bc1h = (half_t*)carve(64 * sizeof(half_t));
    half_t* Wc2h = (half_t*)carve(4096 * sizeof(half_t));
    half_t* bc2h = (half_t*)carve(64 * sizeof(half_t));
    half_t* Wc3h = (half_t*)carve(8192 * sizeof(half_t));
    half_t* bc3h = (half_t*)carve(128 * sizeof(half_t));
    half_t* Wc3T = (half_t*)carve(8192 * sizeof(half_t));
    half_t* Wqh  = (half_t*)carve(65536 * sizeof(half_t));
    half_t* bqh  = (half_t*)carve(256 * sizeof(half_t));
    half_t* WkT  = (half_t*)carve(65536 * sizeof(half_t));
    half_t* bkh  = (half_t*)carve(256 * sizeof(half_t));

    auto cvt = [&](int in_idx, half_t* dst, int n) {
        k_f32_to_f16<<<(n + 255) / 256, 256, 0, stream>>>((const float*)d_in[in_idx], dst, n);
    };
    cvt(6, Wc1h, 576);   cvt(7, bc1h, 64);
    cvt(8, Wc2h, 4096);  cvt(9, bc2h, 64);
    cvt(10, Wc3h, 8192); cvt(11, bc3h, 128);
    cvt(12, Wqh, 65536); cvt(13, bqh, 256);
    cvt(15, bkh, 256);
    // transposed f16 copies: contiguous-K B fragments
    k_f32_to_f16_t<<<(8192 + 255) / 256, 256, 0, stream>>>((const float*)d_in[10], Wc3T, 128, 64);
    k_f32_to_f16_t<<<(65536 + 255) / 256, 256, 0, stream>>>((const float*)d_in[14], WkT, 256, 256);

    const float* x = (const float*)d_in[0];
    k_convs1<<<P / 256, 256, 0, stream>>>(x, (const float*)d_in[2], (const float*)d_in[3],
                                          (const float*)d_in[4], (const float*)d_in[5], fh);
    k_knn<<<P / 256, 256, 0, stream>>>(x, idxb);
    k_point<<<P / 64, 128, 0, stream>>>(x, fh, Wc1h, bc1h, Wc2h, bc2h, Wc3h, bc3h,
                                        Wc3T, Wqh, bqh, WkT, bkh, hout, qbv);
    k_attn<<<P / 8, 256, 0, stream>>>(x, fh, idxb, Wc1h, bc1h, Wc2h, bc2h,
                                      hout, qbv, (float*)d_out);
}
